// BoaBlock_50173807952926
// MI455X (gfx1250) — matrix operand order repaired and checked by reading, not compiled, not run on hardware
//
#include <hip/hip_runtime.h>

// ---------------------------------------------------------------------------
// MI455X (gfx1250) fused BoaBlock kernel, rev 2.
// Roofline: ~32 GFLOP, ~1.3 GB HBM traffic fully fused -> ~55us @ 23.3 TB/s,
// provided (a) all contractions run on v_wmma_f32_16x16x32_f16 and (b) there
// is enough memory-level parallelism. Rev 2 runs 4 waves / workgroup
// (~62 KB LDS -> ~5 WGs/WGP -> 20 waves/WGP) and builds the M-matrix WMMA
// A-fragments directly from global float4 loads (no LDS staging for the
// 512 MB edge_matrices stream; one fragment feeds both overlap types).
// ---------------------------------------------------------------------------

typedef __attribute__((ext_vector_type(16))) _Float16 v16h;
typedef __attribute__((ext_vector_type(8)))  _Float16 v8h;
typedef __attribute__((ext_vector_type(8)))  float    v8f;

static __device__ __forceinline__ v8f wmma16x16x32(v16h a, v16h b, v8f c) {
  // (neg_a, A, neg_b, B, c_mod, C, reuse_a, reuse_b)
  return __builtin_amdgcn_wmma_f32_16x16x32_f16(false, a, false, b, (short)0, c,
                                                false, false);
}

static __device__ __forceinline__ v16h cat16(v8h a, v8h b) {
  v16h r;
#pragma unroll
  for (int i = 0; i < 8; ++i) { r[i] = a[i]; r[8 + i] = b[i]; }
  return r;
}

static __device__ __forceinline__ v16h zero_v16h() {
  v16h r;
#pragma unroll
  for (int i = 0; i < 16; ++i) r[i] = (_Float16)0.0f;
  return r;
}

// Pack four float4 (16 f32) into one f16 fragment (v_cvt_pk_f16_f32 pairs).
static __device__ __forceinline__ v16h cvt16(float4 a, float4 b, float4 c,
                                             float4 d) {
  v16h r;
  r[0] = (_Float16)a.x;  r[1] = (_Float16)a.y;
  r[2] = (_Float16)a.z;  r[3] = (_Float16)a.w;
  r[4] = (_Float16)b.x;  r[5] = (_Float16)b.y;
  r[6] = (_Float16)b.z;  r[7] = (_Float16)b.w;
  r[8] = (_Float16)c.x;  r[9] = (_Float16)c.y;
  r[10] = (_Float16)c.z; r[11] = (_Float16)c.w;
  r[12] = (_Float16)d.x; r[13] = (_Float16)d.y;
  r[14] = (_Float16)d.z; r[15] = (_Float16)d.w;
  return r;
}

static __device__ __forceinline__ v8h lds_v8h(const _Float16* p) {
  return *(const v8h*)p;  // ds_load_b128
}

// Wave-local LDS fence: DS ops are in-order per wave; this pins the compiler
// and drains DScnt so cross-lane LDS data is architecturally visible.
static __device__ __forceinline__ void wave_lds_fence() {
  asm volatile("s_wait_dscnt 0x0" ::: "memory");
}

// Load a 32x8 f32 matrix from global, store its f16 TRANSPOSE (8x32) to LDS.
static __device__ __forceinline__ void load_transpose_32x8(
    const float* __restrict__ src, _Float16 (*dst)[32], int l) {
  const float4* p = (const float4*)src;
#pragma unroll
  for (int it = 0; it < 2; ++it) {
    float4 v = p[l + 32 * it];
    int base = (l + 32 * it) * 4;   // row d = base/8, col c0 = base%8 (0 or 4)
    int d = base >> 3;
    int c0 = base & 7;
    dst[c0 + 0][d] = (_Float16)v.x;
    dst[c0 + 1][d] = (_Float16)v.y;
    dst[c0 + 2][d] = (_Float16)v.z;
    dst[c0 + 3][d] = (_Float16)v.w;
  }
}

// Pre-transpose + f16-convert the MLP weights once per launch (L2-resident).
__global__ void prep_weights(const float* __restrict__ W1,
                             const float* __restrict__ W2,
                             _Float16* __restrict__ W1t,
                             _Float16* __restrict__ W2t) {
  int i = blockIdx.x * blockDim.x + threadIdx.x;
  if (i < 128 * 128) {            // W1t[n][k] = W1[k][n]
    int n = i >> 7, k = i & 127;
    W1t[i] = (_Float16)W1[k * 128 + n];
  }
  if (i < 288 * 128) {            // W2t[n][k] = W2[k][n]
    int n = i >> 7, k = i & 127;
    W2t[i] = (_Float16)W2[k * 288 + n];
  }
}

__global__ __launch_bounds__(128) void boa_fused(
    const float* __restrict__ x, const int* __restrict__ ei,
    const float* __restrict__ efa, const float* __restrict__ efb,
    const float* __restrict__ em, const _Float16* __restrict__ W1t,
    const float* __restrict__ b1g, const float* __restrict__ lng,
    const float* __restrict__ lnb, const _Float16* __restrict__ W2t,
    const float* __restrict__ b2g, float* __restrict__ outA,
    float* __restrict__ outB, int E) {
  // Transposed f16 copies of the four 32x8 per-edge matrices: [arr][edge][c][d]
  // arr: 0 = edge_features_a, 1 = edge_features_b, 2 = x[a], 3 = x[b]
  __shared__ __align__(16) _Float16 sAt[4][16][8][32];     // 32 KB
  __shared__ __align__(16) _Float16 sUt[4][2][2][8][32];   // 8 KB (per-wave U^T)
  __shared__ __align__(16) _Float16 sH[16][128];           // 4 KB   h (f16)
  __shared__ __align__(16) float    sR[16][288];           // 18 KB  f32 staging

  const int tid = threadIdx.x;
  const int l = tid & 31;        // lane in wave (wave32)
  const int w = tid >> 5;        // wave id (0..3)
  const int E0 = blockIdx.x * 16;

  // ---------------- Stage A: gather + transpose feature matrices -----------
  for (int e = w; e < 16; e += 4) {
    int eg = E0 + e; if (eg > E - 1) eg = E - 1;   // tail clamp (uniform)
    int ia = ei[eg];
    int ib = ei[E + eg];
    load_transpose_32x8(efa + (size_t)eg * 256, sAt[0][e], l);
    load_transpose_32x8(efb + (size_t)eg * 256, sAt[1][e], l);
    load_transpose_32x8(x + (size_t)ia * 256, sAt[2][e], l);
    load_transpose_32x8(x + (size_t)ib * 256, sAt[3][e], l);
  }
  __syncthreads();

  // ------- Stage B: overlaps O = A^T (M B); 2 pairs per wave ---------------
  for (int p = w; p < 8; p += 4) {
    const int e0 = 2 * p, e1 = 2 * p + 1;
    wave_lds_fence();            // sUt[w] reuse guard across pair iterations
#pragma unroll
    for (int s = 0; s < 2; ++s) {
      const int e = (s == 0) ? e0 : e1;
      int eg = E0 + e; if (eg > E - 1) eg = E - 1;
      const float* Mp = em + (size_t)eg * 1024;
      if (E0 + e + 4 < E)        // prefetch a later M row-block
        __builtin_prefetch(em + (size_t)(E0 + e + 4) * 1024, 0, 1);
      // B operands (columns of B_e / Xb_e): lane<16 -> K0..7,16..23, else K8..15,24..31
      const int n7 = l & 7;
      const int kb = (l < 16) ? 0 : 8;
      v16h bf0 = cat16(lds_v8h(&sAt[1][e][n7][kb]),
                       lds_v8h(&sAt[1][e][n7][kb + 16]));
      v16h bf1 = cat16(lds_v8h(&sAt[3][e][n7][kb]),
                       lds_v8h(&sAt[3][e][n7][kb + 16]));
#pragma unroll
      for (int r = 0; r < 2; ++r) {          // U row tiles (d = 16r..16r+15)
        // A fragment: rows of M straight from global (f32 -> f16), K chunks
        // {0..7,16..23} for lanes<16, {8..15,24..31} for lanes>=16.
        int row = 16 * r + (l & 15);
        int ka = (l < 16) ? 0 : 8;
        const float* rp = Mp + row * 32 + ka;
        float4 m0 = *(const float4*)(rp);
        float4 m1 = *(const float4*)(rp + 4);
        float4 m2 = *(const float4*)(rp + 16);
        float4 m3 = *(const float4*)(rp + 20);
        v16h af = cvt16(m0, m1, m2, m3);
        v8f acc0 = {}, acc1 = {};
        acc0 = wmma16x16x32(af, bf0, acc0);  // U_edge[16r+m][n]
        acc1 = wmma16x16x32(af, bf1, acc1);  // U_node[16r+m][n]
        int n = l & 15, hi = (l >= 16) ? 8 : 0;
        if (n < 8) {                         // store U^T column chunks
          v8h u0, u1;
#pragma unroll
          for (int vv = 0; vv < 8; ++vv) {
            u0[vv] = (_Float16)acc0[vv];
            u1[vv] = (_Float16)acc1[vv];
          }
          *(v8h*)&sUt[w][s][0][n][16 * r + hi] = u0;
          *(v8h*)&sUt[w][s][1][n][16 * r + hi] = u1;
        }
      }
    }
    wave_lds_fence();            // U stores -> O loads (same wave)
    // O = A^T @ U for both edges of the pair (diagonal 8x8 blocks)
#pragma unroll
    for (int t = 0; t < 2; ++t) {
      const int aArr = (t == 0) ? 0 : 2;
      int lo = l & 15;
      int ePick = (lo < 8) ? e0 : e1;
      int slot = (lo < 8) ? 0 : 1;
      int c = lo & 7;
      int ka = (l < 16) ? 0 : 8;
      v16h af = cat16(lds_v8h(&sAt[aArr][ePick][c][ka]),
                      lds_v8h(&sAt[aArr][ePick][c][ka + 16]));
      int kb = (l < 16) ? 0 : 8;
      v16h bf = cat16(lds_v8h(&sUt[w][slot][t][c][kb]),
                      lds_v8h(&sUt[w][slot][t][c][kb + 16]));
      v8f acc = {};
      acc = wmma16x16x32(af, bf, acc);
      // D[c][m] (edge0: block 0..7) and D[8+c][8+m] (edge1: block 8..15)
      if (l < 8) {
#pragma unroll
        for (int vv = 0; vv < 8; ++vv)
          sH[e0][t * 64 + vv * 8 + l] = (_Float16)acc[vv];
      } else if (l >= 24) {
#pragma unroll
        for (int vv = 0; vv < 8; ++vv)
          sH[e1][t * 64 + vv * 8 + (l - 24)] = (_Float16)acc[vv];
      }
    }
  }
  __syncthreads();

  // ---------------- Stage C: h1 = silu(h @ W1 + b1), 2 N-tiles per wave ----
  for (int nt = w; nt < 8; nt += 4) {
    v8f acc = {};
#pragma unroll
    for (int kt = 0; kt < 4; ++kt) {
      int eRow = l & 15;
      int k0 = kt * 32 + ((l < 16) ? 0 : 8);
      v16h af = cat16(lds_v8h(&sH[eRow][k0]), lds_v8h(&sH[eRow][k0 + 16]));
      int n = nt * 16 + (l & 15);
      const v8h* wp =
          (const v8h*)(W1t + n * 128 + kt * 32 + ((l < 16) ? 0 : 8));
      acc = wmma16x16x32(af, cat16(wp[0], wp[2]), acc);
    }
    int j = nt * 16 + (l & 15);
    float bb = b1g[j];
    int ebase = (l < 16) ? 0 : 8;
#pragma unroll
    for (int vv = 0; vv < 8; ++vv) {
      float h = acc[vv] + bb;
      h = h / (1.0f + __expf(-h));            // silu
      sR[ebase + vv][j] = h;                  // f32 for exact LayerNorm stats
    }
  }
  __syncthreads();

  // ------------- Stage D: LayerNorm, 8 lanes per edge (shfl_xor tree) ------
  {
    int e = tid >> 3, part = tid & 7;         // 16 features per lane
    float sum = 0.f, sq = 0.f;
#pragma unroll
    for (int i = 0; i < 4; ++i) {
      float4 v = *(const float4*)&sR[e][part * 16 + i * 4];
      sum += v.x + v.y + v.z + v.w;
      sq += v.x * v.x + v.y * v.y + v.z * v.z + v.w * v.w;
    }
#pragma unroll
    for (int m = 1; m < 8; m <<= 1) {         // 8-lane groups stay in-wave
      sum += __shfl_xor(sum, m, 32);
      sq += __shfl_xor(sq, m, 32);
    }
    float mu = sum * (1.0f / 128.0f);
    float var = sq * (1.0f / 128.0f) - mu * mu;
    float inv = rsqrtf(var + 1e-5f);
#pragma unroll
    for (int i = 0; i < 2; ++i) {
      int j0 = part * 16 + i * 8;
      v8h hv;
#pragma unroll
      for (int u = 0; u < 8; ++u) {
        int j = j0 + u;
        hv[u] = (_Float16)((sR[e][j] - mu) * inv * lng[j] + lnb[j]);
      }
      *(v8h*)&sH[e][j0] = hv;                 // back to f16 for W2 WMMA
    }
  }
  __syncthreads();

  // ---------------- Stage E: mlp = h @ W2 + b2, 18 N-tiles over 4 waves ----
  for (int nt = w; nt < 18; nt += 4) {
    v8f acc = {};
#pragma unroll
    for (int kt = 0; kt < 4; ++kt) {
      int eRow = l & 15;
      int k0 = kt * 32 + ((l < 16) ? 0 : 8);
      v16h af = cat16(lds_v8h(&sH[eRow][k0]), lds_v8h(&sH[eRow][k0 + 16]));
      int n = nt * 16 + (l & 15);
      const v8h* wp =
          (const v8h*)(W2t + n * 128 + kt * 32 + ((l < 16) ? 0 : 8));
      acc = wmma16x16x32(af, cat16(wp[0], wp[2]), acc);
    }
    int j = nt * 16 + (l & 15);
    float bb = b2g[j];
    int ebase = (l < 16) ? 0 : 8;
#pragma unroll
    for (int vv = 0; vv < 8; ++vv) sR[ebase + vv][j] = acc[vv] + bb;
  }
  __syncthreads();

  // ------- Stage F: Frobenius-normalize 8x8 weights; fold factors & /2 -----
  if (tid < 64) {                 // one lane per (edge, matrix)
    int e = tid & 15, mi = tid >> 4;   // mi: 0=ew_a 1=ew_b 2=nw_a 3=nw_b
    float* wgt = &sR[e][mi * 64];
    float ss = 0.f;
#pragma unroll
    for (int k = 0; k < 16; ++k) {
      float4 v = *(const float4*)(wgt + k * 4);
      ss += v.x * v.x + v.y * v.y + v.z * v.z + v.w * v.w;
    }
    float scale = 0.5f / (sqrtf(ss) + 1e-6f);   // norm + the final /2
    const float* f = &sR[e][256 + mi * 8];      // matching factor block
#pragma unroll
    for (int k = 0; k < 8; ++k)
#pragma unroll
      for (int c = 0; c < 8; ++c) wgt[k * 8 + c] *= scale * f[c];
  }
  __syncthreads();

  // ---- Stage G: out = [A | Xnode] @ [ew_s ; nw_s], 4 edges per wave -------
  for (int e = w; e < 16; e += 4) {
    const bool valid = (E0 + e) < E;
#pragma unroll
    for (int side = 0; side < 2; ++side) {
      const int arr1 = (side == 0) ? 0 : 1;     // A_e  or B_e
      const int arr2 = (side == 0) ? 2 : 3;     // Xa   or Xb
      const int oW1 = (side == 0) ? 0 : 64;     // ew (scaled)
      const int oW2 = (side == 0) ? 128 : 192;  // nw (scaled)
      v16h bf = zero_v16h();
      if ((l & 15) < 8) {                       // B cols 8..15 & K16..31 = 0
#pragma unroll
        for (int k = 0; k < 8; ++k) {
          bf[k] = (_Float16)sR[e][((l < 16) ? oW1 : oW2) + k * 8 + (l & 15)];  // lanes<16: K0..7 = ew rows, lanes>=16: K8..15 = nw rows
        }
      }
      float* outp = ((side == 0) ? outA : outB) + (size_t)(E0 + e) * 256;
#pragma unroll
      for (int r = 0; r < 2; ++r) {
        int d = 16 * r + (l & 15);
        int srcArr = (l < 16) ? arr1 : arr2;    // lanes<16 feed K0..7 (A),
        v16h af = zero_v16h();                  // lanes>=16 feed K8..15 (X)
#pragma unroll
        for (int k = 0; k < 8; ++k) af[k] = sAt[srcArr][e][k][d];
        v8f acc = {};
        acc = wmma16x16x32(af, bf, acc);
        int c = l & 15, hi = (l >= 16) ? 8 : 0;
        if (valid && c < 8) {
#pragma unroll
          for (int vv = 0; vv < 8; ++vv)
            outp[(16 * r + hi + vv) * 8 + c] = acc[vv];
        }
      }
    }
  }
}

extern "C" void kernel_launch(void* const* d_in, const int* in_sizes, int n_in,
                              void* d_out, int out_size, void* d_ws,
                              size_t ws_size, hipStream_t stream) {
  (void)n_in; (void)out_size; (void)ws_size;
  const float* x   = (const float*)d_in[0];
  const int*   ei  = (const int*)d_in[1];
  const float* efa = (const float*)d_in[2];
  const float* efb = (const float*)d_in[3];
  const float* em  = (const float*)d_in[4];
  const float* W1  = (const float*)d_in[5];
  const float* b1  = (const float*)d_in[6];
  const float* lng = (const float*)d_in[7];
  const float* lnb = (const float*)d_in[8];
  const float* W2  = (const float*)d_in[9];
  const float* b2  = (const float*)d_in[10];
  const int E = in_sizes[1] / 2;

  _Float16* W1t = (_Float16*)d_ws;            // 128*128 f16 = 32 KB
  _Float16* W2t = W1t + 128 * 128;            // 288*128 f16 = 72 KB
  float* outA = (float*)d_out;
  float* outB = outA + (size_t)E * 256;

  prep_weights<<<(288 * 128 + 255) / 256, 256, 0, stream>>>(W1, W2, W1t, W2t);
  boa_fused<<<(E + 15) / 16, 128, 0, stream>>>(x, ei, efa, efb, em, W1t, b1,
                                               lng, lnb, W2t, b2, outA, outB,
                                               E);
}
